// DAIE_Model_45354854645964
// MI455X (gfx1250) — matrix operand order repaired and checked by reading, not compiled, not run on hardware
//
#include <hip/hip_runtime.h>
#include <hip/hip_bf16.h>

typedef __attribute__((ext_vector_type(16))) _Float16 v16h;
typedef __attribute__((ext_vector_type(8)))  _Float16 v8h;
typedef __attribute__((ext_vector_type(8)))  float    v8f;

#define BATCH 512
#define NT    64
#define NV    50
#define DIM   300
#define NH    5
#define ET    512
#define EV    600
#define HD    1500      // H*D
#define KP    320       // K padded to multiple of 32
#define NPAD  1536      // HD padded to multiple of 128
#define NGROUPS 12      // NPAD/128
#define MAXE  600
#define MAXN  64

// ---------------- conversion kernels (f32 -> padded f16) ----------------
__global__ void cvt_x_kernel(const float* __restrict__ x, _Float16* __restrict__ xh, long M){
  long idx = (long)blockIdx.x * blockDim.x + threadIdx.x;
  long total = M * KP;
  if (idx >= total) return;
  int  k   = (int)(idx % KP);
  long row = idx / KP;
  float v = (k < DIM) ? x[row * DIM + k] : 0.0f;
  xh[idx] = (_Float16)v;
}

// W[300][1500] -> wT[NPAD][KP] with wT[n][k] = W[k][n], zero padded
__global__ void cvt_w_kernel(const float* __restrict__ W, _Float16* __restrict__ wT){
  int idx = blockIdx.x * blockDim.x + threadIdx.x;
  if (idx >= NPAD * KP) return;
  int k = idx % KP;
  int n = idx / KP;
  float v = (n < HD && k < DIM) ? W[(long)k * HD + n] : 0.0f;
  wT[idx] = (_Float16)v;
}

// ---------------- WMMA GEMM: C[M][1500] = A[M][320] * BT[n][k]^T ----------------
// Each wave computes a 16(M) x 128(N) tile: 8 C-fragments, 10 K-steps of k=32.
// A fragment is loaded once per K-step and reused by all 8 wmma ops.
__global__ void gemm_wmma_kernel(const _Float16* __restrict__ A,
                                 const _Float16* __restrict__ BT,
                                 float* __restrict__ C, int totalWaves){
  int wid = blockIdx.x * (blockDim.x >> 5) + (threadIdx.x >> 5);
  if (wid >= totalWaves) return;
  int lane = threadIdx.x & 31;
  int mT = wid / NGROUPS;
  int nG = wid - mT * NGROUPS;
  int half = lane >> 4;         // K-half selector
  int l16  = lane & 15;

  long arow = (long)(mT * 16 + l16) * KP;
  const _Float16* bp = BT + (long)(nG * 128 + l16) * KP + half * 16;

  v8f acc[8];
  #pragma unroll
  for (int s = 0; s < 8; ++s) acc[s] = (v8f){};

  #pragma unroll
  for (int ks = 0; ks < 10; ++ks){
    int k0 = ks * 32;
    // A fragment: lane holds row m=l16; elems e -> K=k0+half*16+e (contiguous)
    v8h alo = *(const v8h*)(A + arow + k0 + half * 16);
    v8h ahi = *(const v8h*)(A + arow + k0 + half * 16 + 8);
    v16h a = __builtin_shufflevector(alo, ahi, 0,1,2,3,4,5,6,7,8,9,10,11,12,13,14,15);
    #pragma unroll
    for (int s = 0; s < 8; ++s){
      // B fragment: lane holds col n=l16+16*s, elems e -> K=k0+half*16+e (contiguous 32B)
      v16h b = *(const v16h*)(bp + (long)s * 16 * KP + k0);
      acc[s] = __builtin_amdgcn_wmma_f32_16x16x32_f16(false, a, false, b, (short)0, acc[s], false, false);
    }
  }
  int colBase = nG * 128 + l16;
  #pragma unroll
  for (int r = 0; r < 8; ++r){
    long base = (long)(mT * 16 + r + 8 * half) * HD;
    #pragma unroll
    for (int s = 0; s < 8; ++s){
      int c = colBase + s * 16;
      if (c < HD) C[base + c] = acc[s][r];
    }
  }
}

// ---------------- fused GAT layer (one block per batch sample) ----------------
__global__ void gat_kernel(const float* __restrict__ h,            // [B][N][1500]
                           const int*   __restrict__ edge_index,   // src[E], dst[E]
                           int eBatchStride, int E, int N,
                           const float* __restrict__ a_src,        // [5*300]
                           const float* __restrict__ a_dst,
                           const float* __restrict__ bias,         // [300]
                           const unsigned char* __restrict__ mask, // [B][N] or null
                           const float* __restrict__ ln_w,
                           const float* __restrict__ ln_b,
                           float* __restrict__ xout){              // [B][N][300]
  __shared__ float s_e[MAXE * NH];       // edge logits -> attn (also stages a_src|a_dst)
  __shared__ float s_alS[MAXN * NH];
  __shared__ float s_alD[MAXN * NH];
  __shared__ float s_m[MAXN * NH];
  __shared__ float s_s[MAXN * NH];
  __shared__ float s_out[MAXN * DIM];
  __shared__ int   s_src[MAXE];
  __shared__ int   s_dst[MAXE];

  int b = blockIdx.x, tid = threadIdx.x;
  const int* ei = edge_index + (long)b * eBatchStride;
  for (int j = tid; j < E; j += blockDim.x){ s_src[j] = ei[j]; s_dst[j] = ei[E + j]; }
  // stage attention vectors into s_e (1500 + 1500 = 3000 floats, fits exactly)
  for (int i = tid; i < NH * DIM; i += blockDim.x){ s_e[i] = a_src[i]; s_e[NH * DIM + i] = a_dst[i]; }
  __syncthreads();

  // per-(node,head) attention logits
  for (int idx = tid; idx < N * NH; idx += blockDim.x){
    int t = idx / NH, hh = idx - (idx / NH) * NH;
    const float* hr = h + ((long)b * N + t) * HD + hh * DIM;
    float ss = 0.f, dd = 0.f;
    for (int d = 0; d < DIM; ++d){ float hv = hr[d]; ss += hv * s_e[hh * DIM + d]; dd += hv * s_e[NH * DIM + hh * DIM + d]; }
    s_alS[idx] = ss; s_alD[idx] = dd;
  }
  __syncthreads();

  // edge logits + leaky relu
  for (int idx = tid; idx < E * NH; idx += blockDim.x){
    int j = idx / NH, hh = idx - j * NH;
    float v = s_alS[s_src[j] * NH + hh] + s_alD[s_dst[j] * NH + hh];
    s_e[idx] = (v > 0.f) ? v : 0.2f * v;
  }
  __syncthreads();

  // segment max + segment sum per (dst node, head)
  for (int idx = tid; idx < N * NH; idx += blockDim.x){
    int t = idx / NH, hh = idx - (idx / NH) * NH;
    float mm = -__builtin_inff();
    for (int j = 0; j < E; ++j) if (s_dst[j] == t){ float v = s_e[j * NH + hh]; mm = v > mm ? v : mm; }
    float sum = 0.f;
    for (int j = 0; j < E; ++j) if (s_dst[j] == t) sum += __expf(s_e[j * NH + hh] - mm);
    s_m[idx] = mm; s_s[idx] = sum;
  }
  __syncthreads();

  // attn weights (overwrite logits)
  for (int idx = tid; idx < E * NH; idx += blockDim.x){
    int j = idx / NH, hh = idx - j * NH; int t = s_dst[j];
    s_e[idx] = __expf(s_e[idx] - s_m[t * NH + hh]) / s_s[t * NH + hh];
  }
  for (int i = tid; i < N * DIM; i += blockDim.x) s_out[i] = 0.f;
  __syncthreads();

  // aggregation: edges/heads serial, feature dim across lanes -> a given d is
  // always owned by the same thread, so no atomics/races on s_out.
  // Prefetch the next edge's source row (6000B, 64 lanes x 96B stride) while
  // accumulating the current one (global_prefetch_b8, speculative & safe).
  for (int j = 0; j < E; ++j){
    int sj = s_src[j], dj = s_dst[j];
    const float* hb = h + ((long)b * N + sj) * HD;
    if (j + 1 < E && tid < 64){
      const float* hn = h + ((long)b * N + s_src[j + 1]) * HD;
      __builtin_prefetch(hn + tid * 24, 0, 1);
    }
    #pragma unroll
    for (int hh = 0; hh < NH; ++hh){
      float w = s_e[j * NH + hh];
      const float* hr = hb + hh * DIM;
      for (int d = tid; d < DIM; d += blockDim.x) s_out[dj * DIM + d] += w * hr[d];
    }
  }
  __syncthreads();

  // head mean + bias + relu + mask + layernorm
  if (tid < N){
    int t = tid;
    bool mk = (mask != nullptr) && (mask[(long)b * N + t] != 0);
    const float invH = 1.0f / (float)NH;
    float mu = 0.f;
    for (int d = 0; d < DIM; ++d){
      float v = s_out[t * DIM + d] * invH + bias[d];
      v = v > 0.f ? v : 0.f; if (mk) v = 0.f;
      mu += v;
    }
    mu *= (1.0f / DIM);
    float var = 0.f;
    for (int d = 0; d < DIM; ++d){
      float v = s_out[t * DIM + d] * invH + bias[d];
      v = v > 0.f ? v : 0.f; if (mk) v = 0.f;
      float dv = v - mu; var += dv * dv;
    }
    var *= (1.0f / DIM);
    float rstd = rsqrtf(var + 1e-5f);
    float* orow = xout + ((long)b * N + t) * DIM;
    for (int d = 0; d < DIM; ++d){
      float v = s_out[t * DIM + d] * invH + bias[d];
      v = v > 0.f ? v : 0.f; if (mk) v = 0.f;
      orow[d] = (v - mu) * rstd * ln_w[d] + ln_b[d];
    }
  }
}

// ---------------- small fused kernels ----------------
__global__ void pa_token_kernel(const float* __restrict__ t2, const float* __restrict__ w1,
                                const float* __restrict__ b1, const unsigned char* __restrict__ kpm,
                                float* __restrict__ pa){
  int b = blockIdx.x, t = threadIdx.x;   // blockDim = 64
  __shared__ float sl[NT];
  __shared__ float smax, ssum;
  const float* row = t2 + ((long)b * NT + t) * DIM;
  float acc = b1[0];
  for (int d = 0; d < DIM; ++d) acc += row[d] * w1[d];
  if (kpm[(long)b * NT + t]) acc = -__builtin_inff();
  sl[t] = acc;
  __syncthreads();
  if (t == 0){
    float m = -__builtin_inff();
    for (int i = 0; i < NT; ++i) m = sl[i] > m ? sl[i] : m;
    float s = 0.f;
    for (int i = 0; i < NT; ++i) s += __expf(sl[i] - m);
    smax = m; ssum = s;
  }
  __syncthreads();
  pa[(long)b * NT + t] = __expf(sl[t] - smax) / ssum;
}

__global__ void c_kernel(const float* __restrict__ score, const float* __restrict__ t2,
                         float* __restrict__ cbuf){
  long idx = (long)blockIdx.x * blockDim.x + threadIdx.x;
  if (idx >= (long)BATCH * DIM) return;
  int  d = (int)(idx % DIM);
  long b = idx / DIM;
  float acc = 0.f;
  for (int t = 0; t < NT; ++t){ long o = ((long)b * NT + t) * DIM + d; acc += score[o] * t2[o]; }
  cbuf[idx] = acc;
}

// a_1[b,v] = ((sum_t pa[t]*t2[b,t]) . v2[b,v]) / sqrt(D); out[b, v] = a_1 * pv
__global__ void a1_kernel(const float* __restrict__ t2, const float* __restrict__ v2,
                          const float* __restrict__ pa, const float* __restrict__ pv,
                          float* __restrict__ out){
  int b = blockIdx.x, tid = threadIdx.x;
  __shared__ float u[DIM];
  __shared__ float spa[NT];
  for (int i = tid; i < NT; i += blockDim.x) spa[i] = pa[(long)b * NT + i];
  __syncthreads();
  for (int d = tid; d < DIM; d += blockDim.x){
    float acc = 0.f;
    for (int t = 0; t < NT; ++t) acc += spa[t] * t2[((long)b * NT + t) * DIM + d];
    u[d] = acc;
  }
  __syncthreads();
  const float rs = rsqrtf((float)DIM);
  for (int v = tid; v < NV; v += blockDim.x){
    const float* vr = v2 + ((long)b * NV + v) * DIM;
    float acc = 0.f;
    for (int d = 0; d < DIM; ++d) acc += u[d] * vr[d];
    out[(long)b * (2 * NV) + v] = acc * rs * pv[(long)b * NV + v];
  }
}

__global__ void pa_np_kernel(const float* __restrict__ tnp, const float* __restrict__ cbuf,
                             const float* __restrict__ w2, const float* __restrict__ b2,
                             const unsigned char* __restrict__ npm, float* __restrict__ panp){
  int b = blockIdx.x, t = threadIdx.x;   // blockDim = 128
  __shared__ float sl[NT + 1];
  __shared__ float smax, ssum;
  if (t < NT + 1){
    const float* row = (t < NT) ? (tnp + ((long)b * NT + t) * DIM) : (cbuf + (long)b * DIM);
    float acc = b2[0];
    for (int d = 0; d < DIM; ++d) acc += row[d] * w2[d];
    if (npm[(long)b * (NT + 1) + t]) acc = -__builtin_inff();
    sl[t] = acc;
  }
  __syncthreads();
  if (t == 0){
    float m = -__builtin_inff();
    for (int i = 0; i < NT + 1; ++i) m = sl[i] > m ? sl[i] : m;
    float s = 0.f;
    for (int i = 0; i < NT + 1; ++i) s += __expf(sl[i] - m);
    smax = m; ssum = s;
  }
  __syncthreads();
  if (t < NT + 1) panp[(long)b * (NT + 1) + t] = __expf(sl[t] - smax) / ssum;
}

__global__ void a2_kernel(const float* __restrict__ tnp, const float* __restrict__ cbuf,
                          const float* __restrict__ v3, const float* __restrict__ panp,
                          const float* __restrict__ pv, float* __restrict__ out){
  int b = blockIdx.x, tid = threadIdx.x;
  __shared__ float u[DIM];
  __shared__ float spn[NT + 1];
  for (int i = tid; i < NT + 1; i += blockDim.x) spn[i] = panp[(long)b * (NT + 1) + i];
  __syncthreads();
  for (int d = tid; d < DIM; d += blockDim.x){
    float acc = spn[NT] * cbuf[(long)b * DIM + d];
    for (int t = 0; t < NT; ++t) acc += spn[t] * tnp[((long)b * NT + t) * DIM + d];
    u[d] = acc;
  }
  __syncthreads();
  const float rs = rsqrtf((float)DIM);
  for (int v = tid; v < NV; v += blockDim.x){
    const float* vr = v3 + ((long)b * NV + v) * DIM;
    float acc = 0.f;
    for (int d = 0; d < DIM; ++d) acc += u[d] * vr[d];
    out[(long)b * (2 * NV) + NV + v] = acc * rs * pv[(long)b * NV + v];
  }
}

// ---------------- host orchestration ----------------
extern "C" void kernel_launch(void* const* d_in, const int* in_sizes, int n_in,
                              void* d_out, int out_size, void* d_ws, size_t ws_size,
                              hipStream_t stream) {
  (void)in_sizes; (void)n_in; (void)out_size; (void)ws_size;
  const float* t2   = (const float*)d_in[0];
  const float* v2   = (const float*)d_in[1];
  const int*   ei_t = (const int*)d_in[2];
  const unsigned char* gnn_mask = (const unsigned char*)d_in[3];
  const float* score = (const float*)d_in[4];
  const unsigned char* kpm = (const unsigned char*)d_in[5];
  const unsigned char* npm = (const unsigned char*)d_in[6];
  const int*   ei_v = (const int*)d_in[7];
  const float* pv   = (const float*)d_in[8];
  const float* Wt   = (const float*)d_in[9];
  const float* ast  = (const float*)d_in[10];
  const float* adt  = (const float*)d_in[11];
  const float* bt   = (const float*)d_in[12];
  const float* Wv   = (const float*)d_in[13];
  const float* asv  = (const float*)d_in[14];
  const float* adv  = (const float*)d_in[15];
  const float* bv   = (const float*)d_in[16];
  const float* w1   = (const float*)d_in[17];
  const float* b1   = (const float*)d_in[18];
  const float* w2   = (const float*)d_in[19];
  const float* b2   = (const float*)d_in[20];
  const float* lnw  = (const float*)d_in[21];
  const float* lnb  = (const float*)d_in[22];
  float* out = (float*)d_out;

  char* ws = (char*)d_ws;
  size_t off = 0;
  auto carve = [&](size_t bytes) -> char* {
    char* p = ws + off;
    off = (off + bytes + 255) & ~(size_t)255;
    return p;
  };
  _Float16* xh  = (_Float16*)carve((size_t)BATCH * NT * KP * 2);
  _Float16* wT  = (_Float16*)carve((size_t)NPAD * KP * 2);
  float* hbuf   = (float*)carve((size_t)BATCH * NT * HD * 4);
  float* tnpA   = (float*)carve((size_t)BATCH * NT * DIM * 4);
  float* tnpB   = (float*)carve((size_t)BATCH * NT * DIM * 4);
  float* v3A    = (float*)carve((size_t)BATCH * NV * DIM * 4);
  float* v3B    = (float*)carve((size_t)BATCH * NV * DIM * 4);
  float* paTok  = (float*)carve((size_t)BATCH * NT * 4);
  float* paNp   = (float*)carve((size_t)BATCH * (NT + 1) * 4);
  float* cbuf   = (float*)carve((size_t)BATCH * DIM * 4);

  // branch 1: token attention pooling (uses raw t2/v2)
  pa_token_kernel<<<BATCH, 64, 0, stream>>>(t2, w1, b1, kpm, paTok);
  a1_kernel<<<BATCH, 256, 0, stream>>>(t2, v2, paTok, pv, out);
  {
    long total = (long)BATCH * DIM;
    c_kernel<<<(int)((total + 255) / 256), 256, 0, stream>>>(score, t2, cbuf);
  }

  // text GAT stack
  for (int l = 0; l < 2; ++l){
    const float* xin = (l == 0) ? t2 : tnpA;
    float* xo = (l == 0) ? tnpA : tnpB;
    long tot = (long)BATCH * NT * KP;
    cvt_x_kernel<<<(int)((tot + 255) / 256), 256, 0, stream>>>(xin, xh, (long)BATCH * NT);
    cvt_w_kernel<<<(NPAD * KP + 255) / 256, 256, 0, stream>>>(Wt + (long)l * DIM * HD, wT);
    int totalWaves = (BATCH * NT / 16) * NGROUPS;
    gemm_wmma_kernel<<<(totalWaves + 7) / 8, 256, 0, stream>>>(xh, wT, hbuf, totalWaves);
    gat_kernel<<<BATCH, 256, 0, stream>>>(hbuf, ei_t, 2 * ET, ET, NT,
        ast + (long)l * NH * DIM, adt + (long)l * NH * DIM, bt + (long)l * DIM,
        gnn_mask, lnw, lnb, xo);
  }

  // image GAT stack (shared edge index -> batch stride 0, no mask)
  for (int l = 0; l < 2; ++l){
    const float* xin = (l == 0) ? v2 : v3A;
    float* xo = (l == 0) ? v3A : v3B;
    long tot = (long)BATCH * NV * KP;
    cvt_x_kernel<<<(int)((tot + 255) / 256), 256, 0, stream>>>(xin, xh, (long)BATCH * NV);
    cvt_w_kernel<<<(NPAD * KP + 255) / 256, 256, 0, stream>>>(Wv + (long)l * DIM * HD, wT);
    int totalWaves = (BATCH * NV / 16) * NGROUPS;
    gemm_wmma_kernel<<<(totalWaves + 7) / 8, 256, 0, stream>>>(xh, wT, hbuf, totalWaves);
    gat_kernel<<<BATCH, 256, 0, stream>>>(hbuf, ei_v, 0, EV, NV,
        asv + (long)l * NH * DIM, adv + (long)l * NH * DIM, bv + (long)l * DIM,
        nullptr, lnw, lnb, xo);
  }

  // branch 2: node-pool attention over [tnp ; c] against v3
  pa_np_kernel<<<BATCH, 128, 0, stream>>>(tnpB, cbuf, w2, b2, npm, paNp);
  a2_kernel<<<BATCH, 256, 0, stream>>>(tnpB, cbuf, v3B, paNp, pv, out);
}
